// GNNBase_43009802502150
// MI455X (gfx1250) — compile-verified, matrix operand order checked by reading, not run on hardware
//
#include <hip/hip_runtime.h>
#include <hip/hip_bf16.h>

typedef __attribute__((ext_vector_type(16))) _Float16 v16h;
typedef __attribute__((ext_vector_type(8)))  _Float16 v8h;
typedef __attribute__((ext_vector_type(8)))  float    v8f;

#define B_   512
#define N_   32
#define H_   128
#define WSTR 136   // halves per row in f16 LDS tiles (16B aligned, bank-friendly)

// ---------------------------------------------------------------------------
// WMMA fragment load from LDS (f16, 16x16x32 operand layout).
// A operand: row = M (lane&15 selects M within tile), lanes>=16 take K+8.
// B operand: weights stored transposed [N][K]; row = N, same K pattern.
// Per ISA 7.12.2: v[0..3] = K {0..7 | 8..15} packed, v[4..7] = K {16..23 | 24..31}.
// ---------------------------------------------------------------------------
__device__ __forceinline__ v16h ldfrag(const _Float16* p, int row, int kt, int lane) {
    int kb = kt * 32 + ((lane & 16) ? 8 : 0);
    const v8h* q = (const v8h*)__builtin_assume_aligned(p + row * WSTR + kb, 16);
    v8h lo = q[0];   // K base +0..7
    v8h hi = q[2];   // K base +16..23
    return __builtin_shufflevector(lo, hi, 0,1,2,3,4,5,6,7,8,9,10,11,12,13,14,15);
}

// 32x128 = (32x128 f16 A) @ (128x128 f16 W^T-layout) with bias+relu epilogue
// into f32 scratch. wave w owns N-tile w (16 cols), both M-tiles, K-loop of 4.
// Emits 8 v_wmma_f32_16x16x32_f16 per wave.
__device__ __forceinline__ void gemm32x128(const _Float16* A, const _Float16* Wt,
                                           float* scr, const float* bias,
                                           int wave, int lane) {
    v8f acc0 = {}; v8f acc1 = {};
    int lr = lane & 15;
#pragma unroll
    for (int kt = 0; kt < 4; ++kt) {
        v16h bf = ldfrag(Wt, wave * 16 + lr, kt, lane);
        v16h a0 = ldfrag(A, lr,       kt, lane);
        v16h a1 = ldfrag(A, 16 + lr,  kt, lane);
        acc0 = __builtin_amdgcn_wmma_f32_16x16x32_f16(false, a0, false, bf,
                                                      (short)0, acc0, false, false);
        acc1 = __builtin_amdgcn_wmma_f32_16x16x32_f16(false, a1, false, bf,
                                                      (short)0, acc1, false, false);
    }
    int   n  = wave * 16 + lr;
    float bn = bias[n];
    int   mo = (lane & 16) ? 8 : 0;   // C layout: c[k] -> M = k + 8*(lane>=16)
#pragma unroll
    for (int k = 0; k < 8; ++k) {
        float x0 = acc0[k] + bn;
        float x1 = acc1[k] + bn;
        scr[(mo + k) * H_ + n]       = x0 > 0.f ? x0 : 0.f;
        scr[(16 + mo + k) * H_ + n]  = x1 > 0.f ? x1 : 0.f;
    }
}

// LayerNorm of 32 rows x 128 (already relu'd, in f32 scratch) -> f16 mA.
// wave w handles rows 4w..4w+3; 8 lanes per row, 16 elems each; shfl reduce.
__device__ __forceinline__ void ln_store(const float* scr, _Float16* dst,
                                         const float* g, const float* bet,
                                         int wave, int lane) {
    int row = wave * 4 + (lane >> 3);
    int h0  = (lane & 7) * 16;
    float vals[16]; float s = 0.f, ss = 0.f;
#pragma unroll
    for (int i = 0; i < 16; ++i) {
        float v = scr[row * H_ + h0 + i];
        vals[i] = v; s += v; ss += v * v;
    }
#pragma unroll
    for (int m = 1; m < 8; m <<= 1) { s += __shfl_xor(s, m, 32); ss += __shfl_xor(ss, m, 32); }
    float mu  = s * (1.f / H_);
    float inv = rsqrtf(ss * (1.f / H_) - mu * mu + 1e-5f);
#pragma unroll
    for (int i = 0; i < 16; ++i) {
        float o = (vals[i] - mu) * inv * g[h0 + i] + bet[h0 + i];
        dst[row * WSTR + h0 + i] = (_Float16)o;
    }
}

// ---------------------------------------------------------------------------
// Phase 1: per-b pipeline through the 3 pair-MLP layers + mask reduce + logits
// ---------------------------------------------------------------------------
extern "C" __global__ __launch_bounds__(256)
void gnn_phase1(const float* __restrict__ node_obs, const float* __restrict__ adj,
                const int*   __restrict__ agent_id, const float* __restrict__ ent_emb,
                const float* __restrict__ lin1_w,   const float* __restrict__ lin1_b,
                const float* __restrict__ ln1_g,    const float* __restrict__ ln1_b,
                const float* __restrict__ linh_w,   const float* __restrict__ linh_b,
                const float* __restrict__ lnh_g,    const float* __restrict__ lnh_b,
                const float* __restrict__ fc2a_w,   const float* __restrict__ fc2a_b,
                const float* __restrict__ fc2b_w,   const float* __restrict__ fc2b_b,
                float* __restrict__ ws_logits, float* __restrict__ ws_hrow) {
    extern __shared__ char smem[];
    _Float16* wh0   = (_Float16*)smem;             // 128*WSTR  f16  (linh_w[0]^T)
    _Float16* wh1   = wh0 + H_ * WSTR;             // 128*WSTR  f16  (linh_w[1]^T)
    _Float16* w2a   = wh1 + H_ * WSTR;             // 128*WSTR  f16  (fc2a_w^T)
    _Float16* mA    = w2a + H_ * WSTR;             // 32*WSTR   f16  activations
    float* scr      = (float*)(mA + N_ * WSTR);    // 32*128    f32  (also stages lin1_w)
    float* hbuf     = scr  + N_ * H_;              // 32*128    f32  h accumulator
    float* base     = hbuf + N_ * H_;              // 32*128    f32  xj @ W1[:31]
    float* elds     = base + N_ * H_;              // 32*32     f32  e = adj*mask
    float* xj       = elds + N_ * N_;              // 32*32     f32  (padded to 32)
    float* w1last   = xj   + N_ * N_;              // 128       f32  W1[31,:]

    int b = blockIdx.x, tid = threadIdx.x;
    int wave = tid >> 5, lane = tid & 31;

    // ---- stage weights (f16, transposed [n][k]) + per-b inputs ----
    for (int idx = tid; idx < H_ * H_; idx += 256) {
        int k = idx >> 7, n = idx & 127;           // coalesced global reads over n
        wh0[n * WSTR + k] = (_Float16)linh_w[0 * H_ * H_ + k * H_ + n];
        wh1[n * WSTR + k] = (_Float16)linh_w[1 * H_ * H_ + k * H_ + n];
        w2a[n * WSTR + k] = (_Float16)fc2a_w[k * H_ + n];
    }
    for (int idx = tid; idx < N_ * H_; idx += 256) scr[idx] = lin1_w[idx];   // W1 f32
    for (int idx = tid; idx < H_;      idx += 256) w1last[idx] = lin1_w[31 * H_ + idx];
    for (int idx = tid; idx < N_ * N_; idx += 256) {
        int j = idx >> 5, k = idx & 31;
        float v;
        if (k < 15)      v = node_obs[(b * N_ + j) * 16 + k];
        else if (k < 31) {
            int ent = (int)node_obs[(b * N_ + j) * 16 + 15];
            v = ent_emb[ent * 16 + (k - 15)];
        } else           v = 0.f;
        xj[idx] = v;
    }
    for (int idx = tid; idx < N_ * N_; idx += 256) {
        float a = adj[b * N_ * N_ + idx];
        elds[idx] = (a > 0.f && a < 1.0f) ? a : 0.f;   // e; mask == (e > 0)
    }
    for (int idx = tid; idx < N_ * H_; idx += 256) hbuf[idx] = 0.f;
    __syncthreads();

    int row = wave * 4 + (lane >> 3);
    int h0  = (lane & 7) * 16;

    // ---- base[r][h] = sum_{k<31} xj[r][k] * W1[k][h]  (rank-1 trick setup) ----
    {
        float acc[16];
#pragma unroll
        for (int i = 0; i < 16; ++i) acc[i] = 0.f;
        for (int k = 0; k < 31; ++k) {
            float xv = xj[row * 32 + k];
#pragma unroll
            for (int i = 0; i < 16; ++i) acc[i] += xv * scr[k * H_ + h0 + i];
        }
#pragma unroll
        for (int i = 0; i < 16; ++i) base[row * H_ + h0 + i] = acc[i];
    }
    __syncthreads();

    // ---- r loop: 3 layers + masked h accumulation ----
    for (int r = 0; r < N_; ++r) {
        float er = elds[r * N_ + row];
        // layer 1 as rank-1 update: relu(base[r] + e*W1_last + b1)
#pragma unroll
        for (int i = 0; i < 16; ++i) {
            float v = base[r * H_ + h0 + i] + er * w1last[h0 + i] + lin1_b[h0 + i];
            scr[row * H_ + h0 + i] = v > 0.f ? v : 0.f;
        }
        __syncthreads();
        ln_store(scr, mA, ln1_g, ln1_b, wave, lane);
        __syncthreads();
        gemm32x128(mA, wh0, scr, linh_b, wave, lane);
        __syncthreads();
        ln_store(scr, mA, lnh_g, lnh_b, wave, lane);
        __syncthreads();
        gemm32x128(mA, wh1, scr, linh_b + H_, wave, lane);
        __syncthreads();
        ln_store(scr, mA, lnh_g + H_, lnh_b + H_, wave, lane);
        __syncthreads();
        // h[c,:] += mask[r][c] * m[c,:]
        if (er > 0.f) {
#pragma unroll
            for (int i = 0; i < 16; ++i)
                hbuf[row * H_ + h0 + i] += (float)mA[row * WSTR + h0 + i];
        }
        __syncthreads();
    }

    // ---- epilogue: agent h-row to ws; logits = relu(h@fc2a+b) @ fc2b + b ----
    int agent = agent_id[b];
#pragma unroll
    for (int i = 0; i < 16; ++i) {
        float hv = hbuf[row * H_ + h0 + i];
        mA[row * WSTR + h0 + i] = (_Float16)hv;
        if (row == agent) ws_hrow[b * H_ + h0 + i] = hv;
    }
    __syncthreads();
    gemm32x128(mA, w2a, scr, fc2a_b, wave, lane);
    __syncthreads();
    {
        float part = 0.f;
#pragma unroll
        for (int i = 0; i < 16; ++i) part += scr[row * H_ + h0 + i] * fc2b_w[h0 + i];
#pragma unroll
        for (int m = 1; m < 8; m <<= 1) part += __shfl_xor(part, m, 32);
        if ((lane & 7) == 0) ws_logits[b * N_ + row] = part + fc2b_b[0];
    }
}

// ---------------------------------------------------------------------------
// Phase 2: global softmax stats over all 16384 logits (axis=0 softmax)
// ---------------------------------------------------------------------------
extern "C" __global__ __launch_bounds__(256)
void gnn_phase2(const float* __restrict__ logits, float* __restrict__ stats) {
    __shared__ float red[256];
    int t = threadIdx.x;
    float mx = -1e30f;
    for (int i = t; i < B_ * N_; i += 256) mx = fmaxf(mx, logits[i]);
    red[t] = mx; __syncthreads();
    for (int s = 128; s > 0; s >>= 1) { if (t < s) red[t] = fmaxf(red[t], red[t + s]); __syncthreads(); }
    float gmax = red[0]; __syncthreads();
    float sum = 0.f;
    for (int i = t; i < B_ * N_; i += 256) sum += expf(logits[i] - gmax);
    red[t] = sum; __syncthreads();
    for (int s = 128; s > 0; s >>= 1) { if (t < s) red[t] += red[t + s]; __syncthreads(); }
    if (t == 0) { stats[0] = gmax; stats[1] = red[0]; }
}

// ---------------------------------------------------------------------------
// Phase 3: per-b agent row only: fc3 -> LN -> *alpha -> fc4 -> LN -> out
// ---------------------------------------------------------------------------
extern "C" __global__ __launch_bounds__(128)
void gnn_phase3(const int* __restrict__ agent_id, const float* __restrict__ ws_hrow,
                const float* __restrict__ logits, const float* __restrict__ stats,
                const float* __restrict__ fc3_w, const float* __restrict__ fc3_b,
                const float* __restrict__ ln3_g, const float* __restrict__ ln3_b,
                const float* __restrict__ fc4_w, const float* __restrict__ fc4_b,
                const float* __restrict__ ln4_g, const float* __restrict__ ln4_b,
                float* __restrict__ out) {
    __shared__ float sh[128];
    __shared__ float red[128];
    int b = blockIdx.x, t = threadIdx.x;
    sh[t] = ws_hrow[b * H_ + t];
    __syncthreads();
    float acc = fc3_b[t];
    for (int k = 0; k < H_; ++k) acc += sh[k] * fc3_w[k * H_ + t];
    acc = fmaxf(acc, 0.f);
    // LN #1
    red[t] = acc; __syncthreads();
    for (int s = 64; s > 0; s >>= 1) { if (t < s) red[t] += red[t + s]; __syncthreads(); }
    float s1 = red[0]; __syncthreads();
    red[t] = acc * acc; __syncthreads();
    for (int s = 64; s > 0; s >>= 1) { if (t < s) red[t] += red[t + s]; __syncthreads(); }
    float s2 = red[0]; __syncthreads();
    float mu  = s1 * (1.f / H_);
    float inv = rsqrtf(s2 * (1.f / H_) - mu * mu + 1e-5f);
    float v   = (acc - mu) * inv * ln3_g[t] + ln3_b[t];
    int   agent = agent_id[b];
    float alpha = expf(logits[b * N_ + agent] - stats[0]) / stats[1];
    sh[t] = alpha * v;
    __syncthreads();
    float y = fc4_b[t];
    for (int k = 0; k < H_; ++k) y += sh[k] * fc4_w[k * H_ + t];
    y = fmaxf(y, 0.f);
    // LN #2
    red[t] = y; __syncthreads();
    for (int s = 64; s > 0; s >>= 1) { if (t < s) red[t] += red[t + s]; __syncthreads(); }
    float u1 = red[0]; __syncthreads();
    red[t] = y * y; __syncthreads();
    for (int s = 64; s > 0; s >>= 1) { if (t < s) red[t] += red[t + s]; __syncthreads(); }
    float u2 = red[0]; __syncthreads();
    float mu2  = u1 * (1.f / H_);
    float inv2 = rsqrtf(u2 * (1.f / H_) - mu2 * mu2 + 1e-5f);
    out[b * H_ + t] = (y - mu2) * inv2 * ln4_g[t] + ln4_b[t];
}

// ---------------------------------------------------------------------------
extern "C" void kernel_launch(void* const* d_in, const int* in_sizes, int n_in,
                              void* d_out, int out_size, void* d_ws, size_t ws_size,
                              hipStream_t stream) {
    const float* node_obs = (const float*)d_in[0];
    const float* adj      = (const float*)d_in[1];
    const int*   agent_id = (const int*)  d_in[2];
    const float* ent_emb  = (const float*)d_in[3];
    const float* lin1_w   = (const float*)d_in[4];
    const float* lin1_b   = (const float*)d_in[5];
    const float* ln1_g    = (const float*)d_in[6];
    const float* ln1_b    = (const float*)d_in[7];
    const float* linh_w   = (const float*)d_in[8];
    const float* linh_b   = (const float*)d_in[9];
    const float* lnh_g    = (const float*)d_in[10];
    const float* lnh_b    = (const float*)d_in[11];
    const float* fc2a_w   = (const float*)d_in[12];
    const float* fc2a_b   = (const float*)d_in[13];
    const float* fc2b_w   = (const float*)d_in[14];
    const float* fc2b_b   = (const float*)d_in[15];
    const float* fc3_w    = (const float*)d_in[16];
    const float* fc3_b    = (const float*)d_in[17];
    const float* ln3_g    = (const float*)d_in[18];
    const float* ln3_b    = (const float*)d_in[19];
    const float* fc4_w    = (const float*)d_in[20];
    const float* fc4_b    = (const float*)d_in[21];
    const float* ln4_g    = (const float*)d_in[22];
    const float* ln4_b    = (const float*)d_in[23];

    float* ws_logits = (float*)d_ws;                 // 16384 f32
    float* ws_hrow   = ws_logits + B_ * N_;          // 512*128 f32
    float* ws_stats  = ws_hrow + B_ * H_;            // 2 f32

    size_t smem = (size_t)3 * H_ * WSTR * sizeof(_Float16)   // 3 weight tiles
                + (size_t)N_ * WSTR * sizeof(_Float16)        // mA
                + (size_t)3 * N_ * H_ * sizeof(float)         // scr, hbuf, base
                + (size_t)2 * N_ * N_ * sizeof(float)         // elds, xj
                + (size_t)H_ * sizeof(float);                 // w1last

    (void)hipFuncSetAttribute(reinterpret_cast<const void*>(gnn_phase1),
                              hipFuncAttributeMaxDynamicSharedMemorySize, (int)smem);

    gnn_phase1<<<B_, 256, smem, stream>>>(node_obs, adj, agent_id, ent_emb,
                                          lin1_w, lin1_b, ln1_g, ln1_b,
                                          linh_w, linh_b, lnh_g, lnh_b,
                                          fc2a_w, fc2a_b, fc2b_w, fc2b_b,
                                          ws_logits, ws_hrow);
    gnn_phase2<<<1, 256, 0, stream>>>(ws_logits, ws_stats);
    gnn_phase3<<<B_, 128, 0, stream>>>(agent_id, ws_hrow, ws_logits, ws_stats,
                                       fc3_w, fc3_b, ln3_g, ln3_b,
                                       fc4_w, fc4_b, ln4_g, ln4_b,
                                       (float*)d_out);
}
